// StructuralDissimilarity_53111565582833
// MI455X (gfx1250) — matrix operand order repaired and checked by reading, not compiled, not run on hardware
//
#include <hip/hip_runtime.h>

// ---------------------------------------------------------------------------
// StructuralDissimilarity (SSIM, 11x11 valid windows) for gfx1250 / MI455X.
// Box-filter passes are banded matmuls routed through V_WMMA_F32_16X16X32_F16
// with an error-compensated f16 hi/lo split (band matrix is exact in f16).
// ---------------------------------------------------------------------------

typedef __attribute__((ext_vector_type(16))) _Float16 v16h;
typedef __attribute__((ext_vector_type(8)))  float    v8f;

#define IMG_H 512
#define IMG_W 512
#define OUT_HW 502            // 512 - 11 + 1
#define TILES 32              // 16-wide tiles per dimension (covers 0..511)
#define TILES_PER_IMG (TILES * TILES)
#define N_IMGS 24             // 8 batches * 3 channels
#define PARTIALS_PER_BATCH (3 * TILES_PER_IMG)

__device__ __forceinline__ v8f wmma_acc(v16h a, v16h b, v8f c) {
    // 8 args: (neg_a, A, neg_b, B, c_mod, C, reuse_a, reuse_b)
    return __builtin_amdgcn_wmma_f32_16x16x32_f16(
        false, a, false, b, (short)0, c, false, false);
}

__global__ __launch_bounds__(32)
void ssim_tile_kernel(const float* __restrict__ x,
                      const float* __restrict__ y,
                      float* __restrict__ partials) {
    const int wg   = blockIdx.x;
    const int img  = wg / TILES_PER_IMG;
    const int tile = wg % TILES_PER_IMG;
    const int tr   = tile / TILES;
    const int tc   = tile % TILES;
    const int r0   = tr * 16;          // first output row of this tile
    const int c0   = tc * 16;          // first output col of this tile
    const int base = img * (IMG_H * IMG_W);

    const int lane = threadIdx.x;      // 0..31 (wave32)
    const int h    = lane >> 4;        // lane half
    const int m    = lane & 15;        // row/col id within half

    // Horizontal band (B operand, 32x16): element e -> K = (e>=8?16:0)+8h+(e&7), N = m.
    // Band[K][N] = 1 iff N <= K <= N+10  (11-wide box, window start = N).
    v16h hband;
#pragma unroll
    for (int e = 0; e < 16; ++e) {
        int K = ((e >= 8) ? 16 : 0) + 8 * h + (e & 7);
        hband[e] = (_Float16)((K >= m && K <= m + 10) ? 1.0f : 0.0f);
    }
    // Vertical band (A operand, 16x32): element e -> K = 16h+e, M = m.
    v16h vband;
#pragma unroll
    for (int e = 0; e < 16; ++e) {
        int K = 16 * h + e;
        vband[e] = (_Float16)((K >= m && K <= m + 10) ? 1.0f : 0.0f);
    }

    // -------- Horizontal pass: H_q = Q(16x32) x Band(32x16), two row halves --
    v8f Hc[5][2];
#pragma unroll
    for (int ah = 0; ah < 2; ++ah) {
        const int row = r0 + 16 * ah + m;      // A-layout: lane holds row M=m
        float xv[16], yv[16];
#pragma unroll
        for (int e = 0; e < 16; ++e) {
            // 16-bit A layout: element e -> input col offset (e>=8?16:0)+8h+(e&7)
            int col = c0 + ((e >= 8) ? 16 : 0) + 8 * h + (e & 7);
            bool ok = (row < IMG_H) && (col < IMG_W);
            int idx = base + row * IMG_W + col;
            xv[e] = ok ? x[idx] : 0.0f;
            yv[e] = ok ? y[idx] : 0.0f;
        }
#pragma unroll
        for (int q = 0; q < 5; ++q) {
            v16h ahi, alo;
#pragma unroll
            for (int e = 0; e < 16; ++e) {
                float v;
                switch (q) {
                    case 0:  v = xv[e];          break;
                    case 1:  v = yv[e];          break;
                    case 2:  v = xv[e] * xv[e];  break;
                    case 3:  v = yv[e] * yv[e];  break;
                    default: v = xv[e] * yv[e];  break;
                }
                _Float16 hi = (_Float16)v;
                ahi[e] = hi;
                alo[e] = (_Float16)(v - (float)hi);   // exact residual split
            }
            v8f c = {};
            c = wmma_acc(ahi, hband, c);
            c = wmma_acc(alo, hband, c);
            Hc[q][ah] = c;
        }
    }

    // -------- Vertical pass: S_q = VBand(16x32) x H(32x16) --------------------
    // Re-shape H from C/D layout to B layout via lane-half swaps (xor 16).
    v8f S[5];
#pragma unroll
    for (int q = 0; q < 5; ++q) {
        v16h bhi, blo;
#pragma unroll
        for (int r = 0; r < 8; ++r) {
            float c0v = Hc[q][0][r];               // rows (8h)+r   of H half 0
            float c1v = Hc[q][1][r];               // rows 16+(8h)+r of H half 1
            float s0  = __shfl_xor(c0v, 16, 32);
            float s1  = __shfl_xor(c1v, 16, 32);
            // B layout target: element e -> H row K = 16h + e, same column m.
            float eA = h ? s1  : c0v;              // element r    -> row 16h+r
            float eB = h ? c1v : s0;               // element r+8  -> row 16h+8+r
            _Float16 hiA = (_Float16)eA;
            bhi[r]     = hiA;
            blo[r]     = (_Float16)(eA - (float)hiA);
            _Float16 hiB = (_Float16)eB;
            bhi[r + 8] = hiB;
            blo[r + 8] = (_Float16)(eB - (float)hiB);
        }
        v8f s = {};
        s = wmma_acc(vband, bhi, s);
        s = wmma_acc(vband, blo, s);
        S[q] = s;
    }

    // -------- SSIM map + masked accumulation ----------------------------------
    float acc = 0.0f;
#pragma unroll
    for (int r = 0; r < 8; ++r) {
        int row = r0 + 8 * h + r;   // C/D layout: M = 8*half + r
        int col = c0 + m;           //             N = lane % 16
        if (row < OUT_HW && col < OUT_HW) {
            float sx  = S[0][r], sy  = S[1][r];
            float sxx = S[2][r], syy = S[3][r], sxy = S[4][r];
            const float n     = 121.0f;
            const float invn1 = 1.0f / 120.0f;
            const float C1v   = 1.0e-4f;   // (0.01)^2
            const float C2v   = 9.0e-4f;   // (0.03)^2
            float mx  = sx / n;
            float my  = sy / n;
            float vx  = (sxx - n * mx * mx) * invn1;
            float vy  = (syy - n * my * my) * invn1;
            float cov = (sxy - n * mx * my) * invn1;
            float ssim = ((2.0f * mx * my + C1v) * (2.0f * cov + C2v)) /
                         ((mx * mx + my * my + C1v) * (vx + vy + C2v));
            acc += 1.0f - ssim;
        }
    }

    // Wave32 butterfly reduction, lane 0 writes the tile partial.
#pragma unroll
    for (int off = 16; off >= 1; off >>= 1)
        acc += __shfl_xor(acc, off, 32);
    if (lane == 0)
        partials[wg] = acc;
}

// Deterministic fixed-order finalize: sum 3072 tile partials per batch.
__global__ __launch_bounds__(256)
void ssim_finalize_kernel(const float* __restrict__ partials,
                          float* __restrict__ out) {
    __shared__ float sm[256];
    const int b = blockIdx.x;
    const int t = threadIdx.x;
    float s = 0.0f;
    for (int i = t; i < PARTIALS_PER_BATCH; i += 256)
        s += partials[b * PARTIALS_PER_BATCH + i];
    sm[t] = s;
    __syncthreads();
    for (int off = 128; off > 0; off >>= 1) {
        if (t < off) sm[t] += sm[t + off];
        __syncthreads();
    }
    if (t == 0)
        out[b] = sm[0] * (0.5f / (3.0f * 502.0f * 502.0f));
}

extern "C" void kernel_launch(void* const* d_in, const int* in_sizes, int n_in,
                              void* d_out, int out_size, void* d_ws, size_t ws_size,
                              hipStream_t stream) {
    const float* x = (const float*)d_in[0];
    const float* y = (const float*)d_in[1];
    float* out      = (float*)d_out;
    float* partials = (float*)d_ws;   // 24576 floats = 96 KB

    const int nblocks = N_IMGS * TILES_PER_IMG;   // 24576 tiles, 1 wave each
    ssim_tile_kernel<<<nblocks, 32, 0, stream>>>(x, y, partials);
    ssim_finalize_kernel<<<8, 256, 0, stream>>>(partials, out);
}
